// WideAndDeepModel_Controller_hard_5677946765432
// MI455X (gfx1250) — matrix operand order repaired and checked by reading, compiled, not run on hardware
//
#include <hip/hip_runtime.h>
#include <hip/hip_bf16.h>
#include <math.h>

// ---------------- problem constants ----------------
#define BATCH 16384
#define NF    39          // fields
#define NE    16          // embed dim
#define DD    624         // NF*NE
#define DP    640         // DD padded to mult of 32 (WMMA K step)
#define NCP   48          // controller out 39 padded to mult of 16
#define N1    1024
#define N2    512
#define N3    256
#define FIELD_VOCAB 26000
#define BN_EPS 1e-5f

// ---------------- WMMA types ----------------
typedef __attribute__((ext_vector_type(16))) __bf16 v16bf;
typedef __attribute__((ext_vector_type(8)))  float  v8f;

union FragU { uint4 u[2]; v16bf v; };

// =====================================================================
// bf16 WMMA GEMM with N register blocking:
//   Y(rows,Np) = A(rows,Kp) * Bpacked + bias
// One wave computes a 16 x (16*NB) strip: the A fragment is loaded once
// per 32-wide k-step and reused across NB v_wmma_f32_16x16x32_bf16 ops
// (load:wmma ratio (2+2*NB)/NB instead of 4/1).
// A is row-major bf16, row stride Kp (Kp % 32 == 0).
// Bpacked is pre-arranged so each lane reads its 16 bf16 contiguously.
// =====================================================================
template <int NB>
__global__ __launch_bounds__(256) void gemm_bf16_wmma(
    const __bf16* __restrict__ A, const __bf16* __restrict__ Bp,
    const float* __restrict__ bias, float* __restrict__ Y,
    int rows, int Kp, int Np, int Nreal)
{
  int wave = (blockIdx.x * blockDim.x + threadIdx.x) >> 5;
  int lane = threadIdx.x & 31;
  int tilesN = Np >> 4;
  int groups = tilesN / NB;                   // N-strips per row of tiles
  int totalWaves = (rows >> 4) * groups;
  if (wave >= totalWaves) return;
  int tm  = wave / groups;
  int tn0 = (wave - tm * groups) * NB;        // first n-tile of this strip
  int m = lane & 15;
  int h = lane >> 4;

  const __bf16* arow  = A + (size_t)((tm << 4) + m) * Kp;
  const __bf16* bbase = Bp + ((size_t)tn0 * 32 + lane) * 16;
  size_t bstride = (size_t)tilesN * 512;      // elems per k-step of packed B
  int ksteps = Kp >> 5;

  v8f c[NB];
  #pragma unroll
  for (int nb = 0; nb < NB; ++nb) c[nb] = (v8f){};

  for (int ks = 0; ks < ksteps; ++ks) {
    FragU a;
    // A fragment: lane (m,h) needs k = ks*32 + [8h..8h+7] and [16+8h..16+8h+7]
    a.u[0] = *(const uint4*)(arow + (ks << 5) + 8 * h);
    a.u[1] = *(const uint4*)(arow + (ks << 5) + 16 + 8 * h);
    const __bf16* bk = bbase + (size_t)ks * bstride;
    #pragma unroll
    for (int nb = 0; nb < NB; ++nb) {
      FragU b;
      const uint4* bp = (const uint4*)(bk + (size_t)nb * 512);
      b.u[0] = bp[0];
      b.u[1] = bp[1];
      c[nb] = __builtin_amdgcn_wmma_f32_16x16x32_bf16(false, a.v, false, b.v,
                                                      (short)0, c[nb], false, false);
    }
  }
  int rowBase = (tm << 4) + (h << 3);
  #pragma unroll
  for (int nb = 0; nb < NB; ++nb) {
    int n = ((tn0 + nb) << 4) + m;
    float bv = (n < Nreal) ? bias[n] : 0.f;
    #pragma unroll
    for (int v = 0; v < 8; ++v)
      Y[(size_t)(rowBase + v) * Np + n] = c[nb][v] + bv;
  }
}

// =====================================================================
// Pack fp32 weight (K x N row-major) into bf16 WMMA-B fragment order,
// zero-padding to (Kp x Np).
// element i = ((ks*tilesN + tn)*32 + lane)*16 + e
// maps to W[k = ks*32 + 8*(lane>>4) + (e&7) + 16*(e>>3), n = tn*16 + (lane&15)]
// =====================================================================
__global__ void pack_weight(const float* __restrict__ W, __bf16* __restrict__ P,
                            int K, int N, int Kp, int Np)
{
  int i = blockIdx.x * blockDim.x + threadIdx.x;
  if (i >= Kp * Np) return;
  int e    = i & 15;
  int lane = (i >> 4) & 31;
  int rest = i >> 9;
  int tilesN = Np >> 4;
  int ks = rest / tilesN;
  int tn = rest - ks * tilesN;
  int n = (tn << 4) + (lane & 15);
  int k = (ks << 5) + ((lane >> 4) << 3) + (e & 7) + ((e >> 3) << 4);
  float v = (k < K && n < N) ? W[(size_t)k * N + n] : 0.f;
  P[i] = (__bf16)v;
}

// =====================================================================
// Init: zero stats region, linear[b] = lin_bias
// =====================================================================
__global__ void init_kernel(float* __restrict__ stats, int nstats,
                            float* __restrict__ linear,
                            const float* __restrict__ lin_bias)
{
  int i = blockIdx.x * blockDim.x + threadIdx.x;
  if (i < nstats) stats[i] = 0.f;
  if (i < BATCH)  linear[i] = lin_bias[0];
}

// =====================================================================
// Gather: embeddings (float4 loads), per-field sum/sumsq (LDS->global
// atomics), wide-linear accumulation.
// =====================================================================
__global__ __launch_bounds__(256) void gather_kernel(
    const int* __restrict__ x, const float* __restrict__ emb,
    const float* __restrict__ lin, float* __restrict__ embedOut,
    float* __restrict__ fieldSum, float* __restrict__ fieldSum2,
    float* __restrict__ linear)
{
  __shared__ float ls[NF], ls2[NF];
  int t = threadIdx.x;
  if (t < NF) { ls[t] = 0.f; ls2[t] = 0.f; }
  __syncthreads();
  int i = blockIdx.x * blockDim.x + t;
  if (i < BATCH * NF) {
    int b = i / NF, f = i - b * NF;
    long idx = (long)x[i] + (long)f * FIELD_VOCAB;
    const float4* er = (const float4*)(emb + idx * NE);
    float4* out = (float4*)(embedOut + (size_t)i * NE);
    float s = 0.f, s2 = 0.f;
    #pragma unroll
    for (int j = 0; j < 4; ++j) {
      float4 v = er[j]; out[j] = v;
      s  += v.x + v.y + v.z + v.w;
      s2 += v.x*v.x + v.y*v.y + v.z*v.z + v.w*v.w;
    }
    atomicAdd(&ls[f], s); atomicAdd(&ls2[f], s2);
    atomicAdd(&linear[b], lin[idx]);
  }
  __syncthreads();
  if (t < NF) { atomicAdd(&fieldSum[t], ls[t]); atomicAdd(&fieldSum2[t], ls2[t]); }
}

// =====================================================================
// Finalize mean/rstd from sum/sumsq (biased variance, like jnp.var)
// =====================================================================
__global__ void finalize_stats(const float* __restrict__ sum,
                               const float* __restrict__ sum2,
                               float* __restrict__ mean, float* __restrict__ rstd,
                               int n, float count)
{
  int i = blockIdx.x * blockDim.x + threadIdx.x;
  if (i >= n) return;
  float m = sum[i] / count;
  float var = sum2[i] / count - m * m;
  mean[i] = m;
  rstd[i] = rsqrtf(var + BN_EPS);
}

// =====================================================================
// Field BN: normalize embed fp32 in place + emit padded bf16 flat (B,DP)
// =====================================================================
__global__ __launch_bounds__(256) void bn_embed_kernel(
    float* __restrict__ embed, const float* __restrict__ mean,
    const float* __restrict__ rstd, const float* __restrict__ g,
    const float* __restrict__ be, __bf16* __restrict__ flat)
{
  size_t i = (size_t)blockIdx.x * blockDim.x + threadIdx.x;
  if (i >= (size_t)BATCH * DP) return;
  int b = (int)(i / DP), col = (int)(i - (size_t)b * DP);
  if (col < DD) {
    int f = col >> 4;
    size_t src = (size_t)b * DD + col;
    float v = (embed[src] - mean[f]) * rstd[f] * g[f] + be[f];
    embed[src] = v;
    flat[i] = (__bf16)v;
  } else {
    flat[i] = (__bf16)0.f;
  }
}

// =====================================================================
// Column stats: coalesced (consecutive threads = consecutive columns),
// row-chunked over grid.y, atomic partial reduction.
// =====================================================================
__global__ __launch_bounds__(256) void colstats_kernel(
    const float* __restrict__ Y, int N,
    float* __restrict__ sum, float* __restrict__ sum2, int rowsPerChunk)
{
  int c = blockIdx.x * blockDim.x + threadIdx.x;
  if (c >= N) return;
  int r0 = blockIdx.y * rowsPerChunk;
  int r1 = r0 + rowsPerChunk; if (r1 > BATCH) r1 = BATCH;
  float s = 0.f, s2 = 0.f;
  for (int r = r0; r < r1; ++r) {
    float v = Y[(size_t)r * N + c];
    s += v; s2 += v * v;
  }
  atomicAdd(&sum[c], s); atomicAdd(&sum2[c], s2);
}

// =====================================================================
// Controller: BN + ReLU + top-k(20 of 39) -> normalized scatter mask.
// One thread per row; strict '>' selection matches top_k tie-break
// (lowest index first). Sum-normalize selected values.
// =====================================================================
__global__ __launch_bounds__(256) void ctrl_topk_kernel(
    const float* __restrict__ yc, const float* __restrict__ mean,
    const float* __restrict__ rstd, const float* __restrict__ g,
    const float* __restrict__ be, const int* __restrict__ kptr,
    float* __restrict__ mask)
{
  int b = blockIdx.x * blockDim.x + threadIdx.x;
  if (b >= BATCH) return;
  int k = *kptr; if (k > NF) k = NF; if (k < 1) k = 1;
  float w[NF];
  #pragma unroll
  for (int f = 0; f < NF; ++f) {
    float v = (yc[(size_t)b * NCP + f] - mean[f]) * rstd[f] * g[f] + be[f];
    w[f] = fmaxf(v, 0.f);
  }
  unsigned long long sel = 0ull;
  float sum = 0.f;
  for (int it = 0; it < k; ++it) {
    int best = 0; float bv = -1e30f;
    #pragma unroll
    for (int f = 0; f < NF; ++f)
      if (!((sel >> f) & 1ull) && w[f] > bv) { bv = w[f]; best = f; }
    sel |= 1ull << best; sum += bv;
  }
  float inv = 1.f / sum;
  #pragma unroll
  for (int f = 0; f < NF; ++f)
    mask[(size_t)b * NF + f] = ((sel >> f) & 1ull) ? w[f] * inv : 0.f;
}

// =====================================================================
// Gate: flat(b,col) = embedN(b,col) * mask(b,col/16), bf16, zero pad.
// =====================================================================
__global__ __launch_bounds__(256) void mask_apply_kernel(
    const float* __restrict__ embed, const float* __restrict__ mask,
    __bf16* __restrict__ flat)
{
  size_t i = (size_t)blockIdx.x * blockDim.x + threadIdx.x;
  if (i >= (size_t)BATCH * DP) return;
  int b = (int)(i / DP), col = (int)(i - (size_t)b * DP);
  if (col < DD)
    flat[i] = (__bf16)(embed[(size_t)b * DD + col] *
                       mask[(size_t)b * NF + (col >> 4)]);
  else
    flat[i] = (__bf16)0.f;
}

// =====================================================================
// BN + ReLU -> bf16 activations for next GEMM
// =====================================================================
__global__ __launch_bounds__(256) void apply_bn_relu_kernel(
    const float* __restrict__ Y, const float* __restrict__ mean,
    const float* __restrict__ rstd, const float* __restrict__ g,
    const float* __restrict__ be, __bf16* __restrict__ H, int N)
{
  size_t i = (size_t)blockIdx.x * blockDim.x + threadIdx.x;
  if (i >= (size_t)BATCH * N) return;
  int c = (int)(i % N);
  float v = (Y[i] - mean[c]) * rstd[c] * g[c] + be[c];
  H[i] = (__bf16)fmaxf(v, 0.f);
}

// =====================================================================
// Final: BN+ReLU on y3 row, dot with w_out (LDS tree reduce),
// add wide linear + b_out, sigmoid.
// =====================================================================
__global__ __launch_bounds__(256) void final_kernel(
    const float* __restrict__ y3, const float* __restrict__ mean,
    const float* __restrict__ rstd, const float* __restrict__ g,
    const float* __restrict__ be, const float* __restrict__ w_out,
    const float* __restrict__ b_out, const float* __restrict__ linear,
    float* __restrict__ out)
{
  __shared__ float red[256];
  int b = blockIdx.x, t = threadIdx.x;
  float v = y3[(size_t)b * N3 + t];
  v = fmaxf((v - mean[t]) * rstd[t] * g[t] + be[t], 0.f) * w_out[t];
  red[t] = v;
  __syncthreads();
  for (int s = 128; s > 0; s >>= 1) {
    if (t < s) red[t] += red[t + s];
    __syncthreads();
  }
  if (t == 0) {
    float z = red[0] + b_out[0] + linear[b];
    out[b] = 1.f / (1.f + __expf(-z));
  }
}

// =====================================================================
// Host orchestration
// =====================================================================
// workspace layout (byte offsets, all 256-aligned)
#define OFF_EMBED 0u                                   // B*DD fp32   40894464
#define OFF_FLAT  (OFF_EMBED + 40894464u)              // B*DP bf16   20971520
#define OFF_MASK  (OFF_FLAT  + 20971520u)              // B*NF fp32    2555904
#define OFF_YC    (OFF_MASK  +  2555904u)              // B*NCP fp32   3145728
#define OFF_Y     (OFF_YC    +  3145728u)              // B*N1 fp32   67108864 (y1/y2/y3)
#define OFF_H     (OFF_Y     + 67108864u)              // B*N1 bf16   33554432 (h1/h2)
#define OFF_LIN   (OFF_H     + 33554432u)              // B fp32         65536
#define OFF_PC    (OFF_LIN   +    65536u)              // DP*NCP bf16    61440
#define OFF_P1    (OFF_PC    +    61440u)              // DP*N1 bf16   1310720
#define OFF_P2    (OFF_P1    +  1310720u)              // N1*N2 bf16   1048576
#define OFF_P3    (OFF_P2    +  1048576u)              // N2*N3 bf16    262144
#define OFF_STATS (OFF_P3    +   262144u)              // 7516 fp32

// stats region float offsets
#define S_FS   0
#define S_FS2  39
#define S_FM   78
#define S_FR   117
#define S_CS   156
#define S_CS2  204
#define S_CM   252
#define S_CR   300
#define S_1S   348
#define S_1S2  1372
#define S_1M   2396
#define S_1R   3420
#define S_2S   4444
#define S_2S2  4956
#define S_2M   5468
#define S_2R   5980
#define S_3S   6492
#define S_3S2  6748
#define S_3M   7004
#define S_3R   7260
#define STATS_FLOATS 7516

extern "C" void kernel_launch(void* const* d_in, const int* in_sizes, int n_in,
                              void* d_out, int out_size, void* d_ws, size_t ws_size,
                              hipStream_t stream)
{
  const int*   x        = (const int*)  d_in[0];
  const float* emb      = (const float*)d_in[1];
  const float* lin      = (const float*)d_in[2];
  const float* lin_bias = (const float*)d_in[3];
  const float* bn_g     = (const float*)d_in[4];
  const float* bn_b     = (const float*)d_in[5];
  const float* ctrl_w   = (const float*)d_in[6];
  const float* ctrl_b   = (const float*)d_in[7];
  const float* ctrl_g   = (const float*)d_in[8];
  const float* ctrl_be  = (const float*)d_in[9];
  const float* w1 = (const float*)d_in[10]; const float* b1 = (const float*)d_in[11];
  const float* g1 = (const float*)d_in[12]; const float* be1 = (const float*)d_in[13];
  const float* w2 = (const float*)d_in[14]; const float* b2 = (const float*)d_in[15];
  const float* g2 = (const float*)d_in[16]; const float* be2 = (const float*)d_in[17];
  const float* w3 = (const float*)d_in[18]; const float* b3 = (const float*)d_in[19];
  const float* g3 = (const float*)d_in[20]; const float* be3 = (const float*)d_in[21];
  const float* w_out = (const float*)d_in[22];
  const float* b_out = (const float*)d_in[23];
  const int*   kptr  = (const int*)  d_in[24];
  float* out = (float*)d_out;

  char* ws = (char*)d_ws;
  float*  embedN = (float*) (ws + OFF_EMBED);
  __bf16* flat   = (__bf16*)(ws + OFF_FLAT);
  float*  mask   = (float*) (ws + OFF_MASK);
  float*  yc     = (float*) (ws + OFF_YC);
  float*  ybuf   = (float*) (ws + OFF_Y);
  __bf16* hbuf   = (__bf16*)(ws + OFF_H);
  float*  linear = (float*) (ws + OFF_LIN);
  __bf16* Pc     = (__bf16*)(ws + OFF_PC);
  __bf16* P1     = (__bf16*)(ws + OFF_P1);
  __bf16* P2     = (__bf16*)(ws + OFF_P2);
  __bf16* P3     = (__bf16*)(ws + OFF_P3);
  float*  st     = (float*) (ws + OFF_STATS);

  const int T = 256;
  const int rowsPerChunk = 256;           // BATCH/64 row chunks for colstats
  dim3 blk(T);

  // ---- 0. pack weights to bf16 WMMA fragment order ----
  pack_weight<<<(DP*NCP + T-1)/T, blk, 0, stream>>>(ctrl_w, Pc, DD, NF, DP, NCP);
  pack_weight<<<(DP*N1  + T-1)/T, blk, 0, stream>>>(w1, P1, DD, N1, DP, N1);
  pack_weight<<<(N1*N2  + T-1)/T, blk, 0, stream>>>(w2, P2, N1, N2, N1, N2);
  pack_weight<<<(N2*N3  + T-1)/T, blk, 0, stream>>>(w3, P3, N2, N3, N2, N3);

  // ---- 1. init stats + wide linear ----
  init_kernel<<<(BATCH + T-1)/T, blk, 0, stream>>>(st, STATS_FLOATS, linear, lin_bias);

  // ---- 2. gather + field stats + wide linear ----
  gather_kernel<<<(BATCH*NF + T-1)/T, blk, 0, stream>>>(
      x, emb, lin, embedN, st+S_FS, st+S_FS2, linear);
  finalize_stats<<<1, blk, 0, stream>>>(st+S_FS, st+S_FS2, st+S_FM, st+S_FR,
                                        NF, (float)(BATCH*NE));

  // ---- 3. field BN -> normalized fp32 + padded bf16 flat ----
  bn_embed_kernel<<<(BATCH*DP + T-1)/T, blk, 0, stream>>>(
      embedN, st+S_FM, st+S_FR, bn_g, bn_b, flat);

  // ---- 4. controller GEMM (B,640)x(640,48), 16x48 strip per wave ----
  {
    int waves = (BATCH/16) * ((NCP/16)/3);
    gemm_bf16_wmma<3><<<(waves*32 + T-1)/T, blk, 0, stream>>>(
        flat, Pc, ctrl_b, yc, BATCH, DP, NCP, NF);
  }
  colstats_kernel<<<dim3((NCP + T-1)/T, BATCH/rowsPerChunk), blk, 0, stream>>>(
      yc, NCP, st+S_CS, st+S_CS2, rowsPerChunk);
  finalize_stats<<<1, blk, 0, stream>>>(st+S_CS, st+S_CS2, st+S_CM, st+S_CR,
                                        NCP, (float)BATCH);

  // ---- 5. BN+ReLU+topk -> mask; gate embed -> masked bf16 flat (in place) ----
  ctrl_topk_kernel<<<(BATCH + T-1)/T, blk, 0, stream>>>(
      yc, st+S_CM, st+S_CR, ctrl_g, ctrl_be, kptr, mask);
  mask_apply_kernel<<<(BATCH*DP + T-1)/T, blk, 0, stream>>>(embedN, mask, flat);

  // ---- 6. MLP layer 1: (B,640)x(640,1024), 16x64 strip per wave ----
  {
    int waves = (BATCH/16) * ((N1/16)/4);
    gemm_bf16_wmma<4><<<(waves*32 + T-1)/T, blk, 0, stream>>>(
        flat, P1, b1, ybuf, BATCH, DP, N1, N1);
  }
  colstats_kernel<<<dim3((N1 + T-1)/T, BATCH/rowsPerChunk), blk, 0, stream>>>(
      ybuf, N1, st+S_1S, st+S_1S2, rowsPerChunk);
  finalize_stats<<<(N1 + T-1)/T, blk, 0, stream>>>(st+S_1S, st+S_1S2,
                                                   st+S_1M, st+S_1R, N1, (float)BATCH);
  apply_bn_relu_kernel<<<((size_t)BATCH*N1 + T-1)/T, blk, 0, stream>>>(
      ybuf, st+S_1M, st+S_1R, g1, be1, hbuf, N1);

  // ---- 7. MLP layer 2: (B,1024)x(1024,512) ----
  {
    int waves = (BATCH/16) * ((N2/16)/4);
    gemm_bf16_wmma<4><<<(waves*32 + T-1)/T, blk, 0, stream>>>(
        hbuf, P2, b2, ybuf, BATCH, N1, N2, N2);
  }
  colstats_kernel<<<dim3((N2 + T-1)/T, BATCH/rowsPerChunk), blk, 0, stream>>>(
      ybuf, N2, st+S_2S, st+S_2S2, rowsPerChunk);
  finalize_stats<<<(N2 + T-1)/T, blk, 0, stream>>>(st+S_2S, st+S_2S2,
                                                   st+S_2M, st+S_2R, N2, (float)BATCH);
  apply_bn_relu_kernel<<<((size_t)BATCH*N2 + T-1)/T, blk, 0, stream>>>(
      ybuf, st+S_2M, st+S_2R, g2, be2, hbuf, N2);

  // ---- 8. MLP layer 3: (B,512)x(512,256) ----
  {
    int waves = (BATCH/16) * ((N3/16)/4);
    gemm_bf16_wmma<4><<<(waves*32 + T-1)/T, blk, 0, stream>>>(
        hbuf, P3, b3, ybuf, BATCH, N2, N3, N3);
  }
  colstats_kernel<<<dim3((N3 + T-1)/T, BATCH/rowsPerChunk), blk, 0, stream>>>(
      ybuf, N3, st+S_3S, st+S_3S2, rowsPerChunk);
  finalize_stats<<<(N3 + T-1)/T, blk, 0, stream>>>(st+S_3S, st+S_3S2,
                                                   st+S_3M, st+S_3R, N3, (float)BATCH);

  // ---- 9. final BN+ReLU dot + wide + sigmoid ----
  final_kernel<<<BATCH, blk, 0, stream>>>(
      ybuf, st+S_3M, st+S_3R, g3, be3, w_out, b_out, linear, out);
}
